// Portao_22333829939902
// MI455X (gfx1250) — matrix operand order repaired and checked by reading, not compiled, not run on hardware
//
#include <hip/hip_runtime.h>
#include <hip/hip_bf16.h>

// ---------------- problem constants (match reference) ----------------
#define N_TOKENS     65536
#define DIM          2048
#define N_EXPERTS    64
#define TOPK         6
#define N_GROUPS     8
#define TOPK_GROUPS  4

#define TOK_PER_BLOCK 128          // 8 waves x 16 tokens
#define WPAD          2056         // bf16 row stride in LDS (2048 + 8 pad -> kills bank conflicts)
#define LDS_W_BYTES   (N_EXPERTS * WPAD * 2)            // 263168 B
#define LOGIT_STRIDE  65                                 // +1 dword pad -> conflict-free
#define LDS_BYTES     (LDS_W_BYTES + TOK_PER_BLOCK * LOGIT_STRIDE * 4)  // 296448 B <= 320KB

// ---------------- WMMA fragment types ----------------
typedef __attribute__((ext_vector_type(16))) __bf16         v16bf;
typedef __attribute__((ext_vector_type(8)))  __bf16         v8bf;
typedef __attribute__((ext_vector_type(16))) unsigned short v16s;
typedef __attribute__((ext_vector_type(8)))  float          v8f;

// native f32 -> bf16 (RNE); should lower to the hardware v_cvt_pk_bf16 path
__device__ __forceinline__ __bf16 tobf(float f) { return (__bf16)f; }

__global__ void __launch_bounds__(256)
moe_route_kernel(const float* __restrict__ x,
                 const float* __restrict__ peso,
                 float* __restrict__ out_w,
                 int*   __restrict__ out_i) {
  extern __shared__ char smem[];
  __bf16* Wlds  = (__bf16*)smem;                        // bf16 weights [64][WPAD]
  float*  logit = (float*)(smem + LDS_W_BYTES);         // [128][65]

  const int tid = threadIdx.x;

  // ---- Phase 0: stage W (f32, global/L2) -> bf16 LDS image, padded rows ----
  // 64 rows x 256 segments of 8 floats; uniform trip count (no divergence).
  for (int i = tid; i < N_EXPERTS * (DIM / 8); i += 256) {
    const int row = i >> 8;           // DIM/8 == 256 segments per row
    const int seg = i & 255;
    const float4* src = (const float4*)(peso + (size_t)row * DIM + seg * 8);
    const float4 w0 = src[0];
    const float4 w1 = src[1];
    v8bf b;
    b[0] = tobf(w0.x); b[1] = tobf(w0.y); b[2] = tobf(w0.z); b[3] = tobf(w0.w);
    b[4] = tobf(w1.x); b[5] = tobf(w1.y); b[6] = tobf(w1.z); b[7] = tobf(w1.w);
    *(v8bf*)(Wlds + (size_t)row * WPAD + seg * 8) = b;
  }
  __syncthreads();

  // ---- Phase 1: WMMA GEMM. Each wave: 16 tokens x 64 experts ----
  const int wave = tid >> 5;
  const int lane = tid & 31;
  const int n    = lane & 15;   // A: token row within tile; B/C: expert column
  const int h    = lane >> 4;   // half-of-wave selector
  const int tok0 = blockIdx.x * TOK_PER_BLOCK + wave * 16;

  const float* xrow = x + (size_t)(tok0 + n) * DIM;

  v8f acc0 = {}, acc1 = {}, acc2 = {}, acc3 = {};

  // Software-pipelined x loads: fetch chunk k0 before the loop, then inside the
  // loop issue chunk k0+32 while converting/multiplying chunk k0.
  float4 a0, a1, a2, a3;
  {
    const float4* p0 = (const float4*)(xrow + h * 16);
    const float4* p1 = (const float4*)(xrow + h * 16 + 8);
    a0 = p0[0]; a1 = p0[1]; a2 = p1[0]; a3 = p1[1];
  }

#pragma unroll 2
  for (int k0 = 0; k0 < DIM; k0 += 32) {
    // prefetch the x stream a few cachelines ahead (global_prefetch_b8)
    __builtin_prefetch(xrow + k0 + 256, 0, 1);

    // issue next chunk's loads now (wrap-indexed on last iter: in-bounds, unused)
    const int kn = (k0 + 32) & (DIM - 1);
    const float4* q0 = (const float4*)(xrow + kn + h * 16);
    const float4* q1 = (const float4*)(xrow + kn + h * 16 + 8);
    const float4 n0 = q0[0], n1 = q0[1], n2 = q1[0], n3 = q1[1];

    // A fragment, this code's own k order:
    //   lane half h: K in [h*16, h*16+16) contiguous (matches B)
    v16bf A;
    A[0]  = tobf(a0.x); A[1]  = tobf(a0.y); A[2]  = tobf(a0.z); A[3]  = tobf(a0.w);
    A[4]  = tobf(a1.x); A[5]  = tobf(a1.y); A[6]  = tobf(a1.z); A[7]  = tobf(a1.w);
    A[8]  = tobf(a2.x); A[9]  = tobf(a2.y); A[10] = tobf(a2.z); A[11] = tobf(a2.w);
    A[12] = tobf(a3.x); A[13] = tobf(a3.y); A[14] = tobf(a3.z); A[15] = tobf(a3.w);

    // B fragments, same k order as A: lane h*16+n holds K = h*16 .. h*16+15
    // of column n -> 32 contiguous bytes of bf16 row (expert e0+n) in LDS.
    const __bf16* wb = Wlds + (size_t)n * WPAD + k0 + h * 16;
    const v16bf B0 = *(const v16bf*)(wb);
    const v16bf B1 = *(const v16bf*)(wb + 16 * WPAD);
    const v16bf B2 = *(const v16bf*)(wb + 32 * WPAD);
    const v16bf B3 = *(const v16bf*)(wb + 48 * WPAD);

    acc0 = __builtin_amdgcn_wmma_f32_16x16x32_bf16(false, A, false, B0, (short)0, acc0, false, false);
    acc1 = __builtin_amdgcn_wmma_f32_16x16x32_bf16(false, A, false, B1, (short)0, acc1, false, false);
    acc2 = __builtin_amdgcn_wmma_f32_16x16x32_bf16(false, A, false, B2, (short)0, acc2, false, false);
    acc3 = __builtin_amdgcn_wmma_f32_16x16x32_bf16(false, A, false, B3, (short)0, acc3, false, false);

    a0 = n0; a1 = n1; a2 = n2; a3 = n3;
  }

  // C/D layout: VGPR j, lanes 0-15 -> M=j, lanes 16-31 -> M=j+8; N = lane&15
  {
    const int wm0 = wave * 16;
#pragma unroll
    for (int j = 0; j < 8; ++j) {
      const int r = (wm0 + j + 8 * h) * LOGIT_STRIDE + n;
      logit[r]      = acc0[j];
      logit[r + 16] = acc1[j];
      logit[r + 32] = acc2[j];
      logit[r + 48] = acc3[j];
    }
  }
  __syncthreads();

  // ---- Phase 2: softmax + group-limited top-k, one thread per token ----
  if (tid < TOK_PER_BLOCK) {
    const float* s = logit + tid * LOGIT_STRIDE;   // bank-conflict-free (stride 65)

    float mx = s[0];
#pragma unroll
    for (int e = 1; e < N_EXPERTS; ++e) mx = fmaxf(mx, s[e]);
    float sum = 0.0f;
#pragma unroll
    for (int e = 0; e < N_EXPERTS; ++e) sum += __expf(s[e] - mx);
    const float inv = 1.0f / sum;                  // ROUTE_SCALE == 1.0

    // group maxima (softmax is monotonic -> selecting on logits == on scores)
    float gs[N_GROUPS];
#pragma unroll
    for (int g = 0; g < N_GROUPS; ++g) {
      float m = s[g * 8];
#pragma unroll
      for (int j = 1; j < 8; ++j) m = fmaxf(m, s[g * 8 + j]);
      gs[g] = m;
    }

    // top-4 groups; ascending scan with strict '>' reproduces top_k tie rule
    unsigned keep = 0;
    for (int i = 0; i < TOPK_GROUPS; ++i) {
      int bg = 0; float bv = -__builtin_inff();
      for (int g = 0; g < N_GROUPS; ++g)
        if (!((keep >> g) & 1u) && gs[g] > bv) { bv = gs[g]; bg = g; }
      keep |= 1u << bg;
    }

    // top-6 experts among kept groups
    unsigned long long chosen = 0ull;
    const size_t gt = (size_t)blockIdx.x * TOK_PER_BLOCK + tid;
    for (int i = 0; i < TOPK; ++i) {
      int be = 0; float bv = -__builtin_inff();
      for (int g = 0; g < N_GROUPS; ++g) {
        if (!((keep >> g) & 1u)) continue;
        for (int j = 0; j < 8; ++j) {
          const int e = g * 8 + j;
          const float v = s[e];
          if (!((chosen >> e) & 1ull) && v > bv) { bv = v; be = e; }
        }
      }
      chosen |= 1ull << be;
      out_w[gt * TOPK + i] = __expf(s[be] - mx) * inv;  // original softmax score
      out_i[gt * TOPK + i] = be;
    }
  }
}

extern "C" void kernel_launch(void* const* d_in, const int* in_sizes, int n_in,
                              void* d_out, int out_size, void* d_ws, size_t ws_size,
                              hipStream_t stream) {
  (void)in_sizes; (void)n_in; (void)out_size; (void)d_ws; (void)ws_size;
  const float* x    = (const float*)d_in[0];
  const float* peso = (const float*)d_in[1];
  float* out_w = (float*)d_out;                              // pesos [N, TOPK] f32
  int*   out_i = (int*)d_out + (size_t)N_TOKENS * TOPK;      // idx   [N, TOPK] i32

  (void)hipFuncSetAttribute((const void*)moe_route_kernel,
                            hipFuncAttributeMaxDynamicSharedMemorySize, LDS_BYTES);
  moe_route_kernel<<<N_TOKENS / TOK_PER_BLOCK, 256, LDS_BYTES, stream>>>(
      x, peso, out_w, out_i);
}
